// TSESC_3461743641144
// MI455X (gfx1250) — matrix operand order repaired and checked by reading, not compiled, not run on hardware
//
#include <hip/hip_runtime.h>
#include <math.h>

// ---------------------------------------------------------------------------
// MI455X (gfx1250) implementation. wave32; WMMA bf16 16x16x32 w/ f32 accum.
// ---------------------------------------------------------------------------

typedef __attribute__((ext_vector_type(16))) __bf16 v16bf;
typedef __attribute__((ext_vector_type(8)))  float  v8f;

#define H_STEPS 12
#define NNODE   10000
#define BGRAPH  4
#define NBT     40000      // nodes per step (B*N)
#define DM      64
#define NEDGE   640000
#define NREL    3
#define NBASE   2
#define PFD     256
#define MAXLEN  96

static __device__ __forceinline__ v8f v8f_zero() {
  v8f z;
#pragma unroll
  for (int i = 0; i < 8; i++) z[i] = 0.f;
  return z;
}

static __device__ __forceinline__ v8f wmma_bf16(v16bf a, v16bf b, v8f c) {
  return __builtin_amdgcn_wmma_f32_16x16x32_bf16(false, a, false, b, (short)0, c,
                                                 false, false);
}

// A-fragment (16x32 bf16) from an LDS row-major bf16 tile of width `width`.
// lane L: row m = L%16, group g = L/16; elements e<8 -> K = kt*32+g*8+e,
// e>=8 -> K = kt*32+16+g*8+(e-8).  (CDNA5 ISA 7.12.2)
static __device__ __forceinline__ v16bf load_afrag(const __bf16* t, int lane,
                                                   int kt, int width) {
  int m = lane & 15, g = lane >> 4;
  const __bf16* p0 = t + m * width + kt * 32 + g * 8;
  v16bf a;
#pragma unroll
  for (int e = 0; e < 8; e++) { a[e] = p0[e]; a[e + 8] = p0[16 + e]; }
  return a;
}

// B-fragment from packed global weights: [(kt*NT+nt)*32 + lane] * 16 halfs.
static __device__ __forceinline__ v16bf load_bfrag(const __bf16* pk, int NT,
                                                   int kt, int nt, int lane) {
  const v16bf* p =
      (const v16bf*)(pk + (size_t)((kt * NT + nt) * 32 + lane) * 16);
  return *p;
}

// Stage a 16x64 f32 tile (row-major, ld=64) into LDS as bf16, optional relu.
static __device__ __forceinline__ void stage_tile(const float* src, long row0,
                                                  long nrows, __bf16* dst,
                                                  int lane, int relu) {
  for (int i = lane; i < 16 * 64; i += 32) {
    int r = i >> 6, c = i & 63;
    long rr = row0 + r;
    float v = (rr < nrows) ? src[rr * 64L + c] : 0.f;
    if (relu && v < 0.f) v = 0.f;
    dst[i] = (__bf16)v;
  }
}

// ---------------------------------------------------------------------------
// Small prep kernels
// ---------------------------------------------------------------------------

__global__ void k_pe(float* pe) {
  int t = blockIdx.x * blockDim.x + threadIdx.x;
  if (t >= MAXLEN * DM) return;
  int pos = t >> 6, d = t & 63;
  int i2 = d & ~1;
  float div = __expf((float)i2 * (-logf(10000.f) / (float)DM));
  float ang = (float)pos * div;
  pe[t] = (d & 1) ? __cosf(ang) : __sinf(ang);
}

// W[r,i,o] = sum_b comp[r,b]*basis[b,i,o]
__global__ void k_combine(const float* basis, const float* comp, int in_d,
                          float* W) {
  int t = blockIdx.x * blockDim.x + threadIdx.x;
  int total = NREL * in_d * DM;
  if (t >= total) return;
  int r = t / (in_d * DM);
  int io = t - r * (in_d * DM);
  W[t] = comp[r * NBASE] * basis[io] + comp[r * NBASE + 1] * basis[in_d * DM + io];
}

// Pack f32 weight matrix (GEMM view B[K][N]) into bf16 B-fragments.
// transpose=0: B[k][n] = W[k*N+n];  transpose=1: B[k][n] = W[n*K+k].
__global__ void k_pack(const float* W, int K, int Nn, int transpose,
                       __bf16* dst) {
  int t = blockIdx.x * blockDim.x + threadIdx.x;
  int total = (K / 32) * (Nn / 16) * 32;
  if (t >= total) return;
  int lane = t & 31, fi = t >> 5;
  int NT = Nn / 16, kt = fi / NT, nt = fi - kt * NT;
  int n = nt * 16 + (lane & 15), g = lane >> 4;
  v16bf v;
#pragma unroll
  for (int e = 0; e < 16; e++) {
    int k = kt * 32 + (e < 8 ? g * 8 + e : 16 + g * 8 + (e - 8));
    float w = transpose ? W[(size_t)n * K + k] : W[(size_t)k * Nn + n];
    v[e] = (__bf16)w;
  }
  *((v16bf*)(dst + (size_t)t * 16)) = v;
}

// ---------------------------------------------------------------------------
// RGCN
// ---------------------------------------------------------------------------

// hop0 root: y[n,c] = x[n,0]*root[0,c] + x[n,1]*root[1,c]   (in_d = 2)
__global__ void k_hop0_root(const float* x, const float* root, float* y) {
  unsigned t = blockIdx.x * blockDim.x + threadIdx.x;
  if (t >= (unsigned)NBT * DM) return;
  unsigned n = t >> 6; int c = (int)(t & 63);
  y[t] = x[n * 2] * root[c] + x[n * 2 + 1] * root[DM + c];
}

// hop0 edges (fused per-edge GEMM, in_d=2): y[dst] += x[src] @ W0[rel]
__global__ void k_hop0_edges(const float* x, const int* src, const int* dst,
                             const int* rel, const float* W0, float* y) {
  unsigned long t = (unsigned long)blockIdx.x * blockDim.x + threadIdx.x;
  if (t >= (unsigned long)NEDGE * DM) return;
  unsigned e = (unsigned)(t >> 6); int c = (int)(t & 63);
  int s = src[e], d = dst[e], r = rel[e];
  float m = x[(unsigned)s * 2] * W0[(r * 2 + 0) * DM + c] +
            x[(unsigned)s * 2 + 1] * W0[(r * 2 + 1) * DM + c];
  atomicAdd(&y[(unsigned)d * DM + c], m);
}

// hop1 edges: out[dst] += h[rel][src]   (h precomputed per relation by WMMA)
__global__ void k_hop1_edges(const int* src, const int* dst, const int* rel,
                             const float* h, float* out) {
  unsigned long t = (unsigned long)blockIdx.x * blockDim.x + threadIdx.x;
  if (t >= (unsigned long)NEDGE * DM) return;
  unsigned e = (unsigned)(t >> 6); int c = (int)(t & 63);
  int s = src[e], d = dst[e], r = rel[e];
  atomicAdd(&out[(unsigned)d * DM + c],
            h[((unsigned long)r * NBT + (unsigned)s) * DM + c]);
}

// ---------------------------------------------------------------------------
// WMMA GEMM: out[M,64] = (relu?)A[M,64] @ W(packed bf16) (+ PE epilogue)
// 6 waves/block, 16 rows per wave.  Used only for per-step M=NBT (fits u32).
// ---------------------------------------------------------------------------
__global__ __launch_bounds__(192) void k_gemm64(const float* A, int nrows,
                                                const __bf16* pk, float* out,
                                                int relu_in, const float* pe,
                                                const int* tidx, int step) {
  __shared__ __bf16 sA[6][16 * 64];
  __shared__ float sO[6][16 * 64];
  int w = threadIdx.x >> 5, lane = threadIdx.x & 31;
  int row0 = blockIdx.x * 96 + w * 16;

  stage_tile(A, row0, nrows, sA[w], lane, relu_in);
  __syncthreads();

  v16bf a0 = load_afrag(sA[w], lane, 0, 64);
  v16bf a1 = load_afrag(sA[w], lane, 1, 64);
  v8f acc[4];
#pragma unroll
  for (int nt = 0; nt < 4; nt++) {
    acc[nt] = v8f_zero();
    acc[nt] = wmma_bf16(a0, load_bfrag(pk, 4, 0, nt, lane), acc[nt]);
    acc[nt] = wmma_bf16(a1, load_bfrag(pk, 4, 1, nt, lane), acc[nt]);
  }
  {
    int n = lane & 15, half = lane >> 4;
#pragma unroll
    for (int nt = 0; nt < 4; nt++)
#pragma unroll
      for (int v = 0; v < 8; v++)
        sO[w][(v + 8 * half) * 64 + nt * 16 + n] = acc[nt][v];
  }
  __syncthreads();
  for (int i = lane; i < 16 * 64; i += 32) {
    int r = i >> 6, c = i & 63;
    unsigned rr = (unsigned)(row0 + r);
    if (rr < (unsigned)nrows) {
      float v = sO[w][i];
      if (pe) {
        // 32-bit division: rr < 2^31, lowers to one 32-bit magic multiply
        unsigned b = rr / (unsigned)NNODE;
        v += pe[tidx[step * BGRAPH + (int)b] * DM + c];
      }
      out[(size_t)rr * 64 + c] = v;
    }
  }
}

// ---------------------------------------------------------------------------
// Fused attention block: o = LN(Xq + fc(softmax(q*k/8)*v)) ; row-local.
// q from Xq, k/v from Xkv (self: Xq==Xkv).
// ---------------------------------------------------------------------------
__global__ __launch_bounds__(192) void k_attn(const float* Xq, const float* Xkv,
                                              long nrows, const __bf16* pkQ,
                                              const __bf16* pkK,
                                              const __bf16* pkV,
                                              const __bf16* pkFC,
                                              const float* fc_b,
                                              const float* ln_g,
                                              const float* ln_b, float* out) {
  __shared__ __bf16 sA[6][16 * 64];  // q-source bf16, later o bf16
  __shared__ __bf16 sB[6][16 * 64];  // kv-source bf16 (cross only)
  __shared__ float sF[6][16 * 64];   // q*k scores, later LN staging
  __shared__ float sMean[6][16], sInv[6][16];
  int w = threadIdx.x >> 5, lane = threadIdx.x & 31;
  long row0 = (long)blockIdx.x * 96 + (long)w * 16;
  bool self = (Xq == Xkv);

  stage_tile(Xq, row0, nrows, sA[w], lane, 0);
  if (!self) stage_tile(Xkv, row0, nrows, sB[w], lane, 0);
  __syncthreads();

  v16bf aq0 = load_afrag(sA[w], lane, 0, 64);
  v16bf aq1 = load_afrag(sA[w], lane, 1, 64);
  const __bf16* kvt = self ? sA[w] : sB[w];
  v16bf ak0 = load_afrag(kvt, lane, 0, 64);
  v16bf ak1 = load_afrag(kvt, lane, 1, 64);

  v8f qa[4], ka[4], va[4];
#pragma unroll
  for (int nt = 0; nt < 4; nt++) {
    qa[nt] = v8f_zero(); ka[nt] = v8f_zero(); va[nt] = v8f_zero();
    qa[nt] = wmma_bf16(aq0, load_bfrag(pkQ, 4, 0, nt, lane), qa[nt]);
    qa[nt] = wmma_bf16(aq1, load_bfrag(pkQ, 4, 1, nt, lane), qa[nt]);
    ka[nt] = wmma_bf16(ak0, load_bfrag(pkK, 4, 0, nt, lane), ka[nt]);
    ka[nt] = wmma_bf16(ak1, load_bfrag(pkK, 4, 1, nt, lane), ka[nt]);
    va[nt] = wmma_bf16(ak0, load_bfrag(pkV, 4, 0, nt, lane), va[nt]);
    va[nt] = wmma_bf16(ak1, load_bfrag(pkV, 4, 1, nt, lane), va[nt]);
  }
  int n = lane & 15, half = lane >> 4;
#pragma unroll
  for (int nt = 0; nt < 4; nt++)
#pragma unroll
    for (int v = 0; v < 8; v++)
      sF[w][(v + 8 * half) * 64 + nt * 16 + n] = qa[nt][v] * ka[nt][v] * 0.125f;
  __syncthreads();

  // softmax over each head's 16 values (heads == 16-col groups)
  if (lane < 16) {
    for (int h = 0; h < 4; h++) {
      float* row = sF[w] + lane * 64 + h * 16;
      float mx = row[0];
#pragma unroll
      for (int j = 1; j < 16; j++) mx = fmaxf(mx, row[j]);
      float s = 0.f;
#pragma unroll
      for (int j = 0; j < 16; j++) { float e = __expf(row[j] - mx); row[j] = e; s += e; }
      float inv = 1.f / s;
#pragma unroll
      for (int j = 0; j < 16; j++) row[j] *= inv;
    }
  }
  __syncthreads();

  // o = v * s ; overwrite sA with o (bf16) for the fc GEMM
#pragma unroll
  for (int nt = 0; nt < 4; nt++)
#pragma unroll
    for (int v = 0; v < 8; v++) {
      int idx = (v + 8 * half) * 64 + nt * 16 + n;
      sA[w][idx] = (__bf16)(va[nt][v] * sF[w][idx]);
    }
  __syncthreads();

  v16bf ao0 = load_afrag(sA[w], lane, 0, 64);
  v16bf ao1 = load_afrag(sA[w], lane, 1, 64);
  v8f oa[4];
#pragma unroll
  for (int nt = 0; nt < 4; nt++) {
    oa[nt] = v8f_zero();
    oa[nt] = wmma_bf16(ao0, load_bfrag(pkFC, 4, 0, nt, lane), oa[nt]);
    oa[nt] = wmma_bf16(ao1, load_bfrag(pkFC, 4, 1, nt, lane), oa[nt]);
  }
  // + fc_b + residual -> sF
#pragma unroll
  for (int nt = 0; nt < 4; nt++)
#pragma unroll
    for (int v = 0; v < 8; v++) {
      long rr = row0 + v + 8 * half;
      int c = nt * 16 + n;
      float res = (rr < nrows) ? Xq[rr * 64L + c] : 0.f;
      sF[w][(v + 8 * half) * 64 + c] = oa[nt][v] + fc_b[c] + res;
    }
  __syncthreads();
  if (lane < 16) {
    float* row = sF[w] + lane * 64;
    float m = 0.f;
#pragma unroll
    for (int j = 0; j < 64; j++) m += row[j];
    m *= (1.f / 64.f);
    float var = 0.f;
#pragma unroll
    for (int j = 0; j < 64; j++) { float d = row[j] - m; var += d * d; }
    var *= (1.f / 64.f);
    sMean[w][lane] = m;
    sInv[w][lane] = rsqrtf(var + 1e-5f);
  }
  __syncthreads();
  for (int i = lane; i < 16 * 64; i += 32) {
    int r = i >> 6, c = i & 63;
    long rr = row0 + r;
    if (rr < nrows)
      out[rr * 64L + c] =
          (sF[w][i] - sMean[w][r]) * sInv[w][r] * ln_g[c] + ln_b[c];
  }
}

// ---------------------------------------------------------------------------
// Fused FF block: out = LN(X + c2(relu(c1(X)+b1))+b2); 256-wide intermediate
// kept in registers/LDS (never hits HBM).
// ---------------------------------------------------------------------------
__global__ __launch_bounds__(192) void k_ff(const float* X, long nrows,
                                            const __bf16* pkC1,
                                            const __bf16* pkC2,
                                            const float* c1_b,
                                            const float* c2_b,
                                            const float* ln_g,
                                            const float* ln_b, float* out) {
  __shared__ __bf16 sA[6][16 * 64];
  __shared__ __bf16 sI[6][16 * 32];
  __shared__ float sF[6][16 * 64];
  __shared__ float sMean[6][16], sInv[6][16];
  int w = threadIdx.x >> 5, lane = threadIdx.x & 31;
  long row0 = (long)blockIdx.x * 96 + (long)w * 16;
  int n = lane & 15, half = lane >> 4;

  stage_tile(X, row0, nrows, sA[w], lane, 0);
  __syncthreads();
  v16bf a0 = load_afrag(sA[w], lane, 0, 64);
  v16bf a1 = load_afrag(sA[w], lane, 1, 64);

  v8f oacc[4];
#pragma unroll
  for (int nt = 0; nt < 4; nt++) oacc[nt] = v8f_zero();

  for (int ch = 0; ch < 8; ch++) {  // 8 chunks of 32 intermediate channels
    v8f y[2];
#pragma unroll
    for (int j = 0; j < 2; j++) {
      int nt = ch * 2 + j;  // NT for c1 = 16
      y[j] = v8f_zero();
      y[j] = wmma_bf16(a0, load_bfrag(pkC1, 16, 0, nt, lane), y[j]);
      y[j] = wmma_bf16(a1, load_bfrag(pkC1, 16, 1, nt, lane), y[j]);
    }
#pragma unroll
    for (int j = 0; j < 2; j++)
#pragma unroll
      for (int v = 0; v < 8; v++) {
        float t = y[j][v] + c1_b[ch * 32 + j * 16 + n];
        t = t > 0.f ? t : 0.f;
        sI[w][(v + 8 * half) * 32 + j * 16 + n] = (__bf16)t;
      }
    __syncthreads();
    v16bf ai = load_afrag(sI[w], lane, 0, 32);
#pragma unroll
    for (int nt = 0; nt < 4; nt++)
      oacc[nt] = wmma_bf16(ai, load_bfrag(pkC2, 4, ch, nt, lane), oacc[nt]);
    __syncthreads();
  }
  // + c2_b + residual -> sF, then LN
#pragma unroll
  for (int nt = 0; nt < 4; nt++)
#pragma unroll
    for (int v = 0; v < 8; v++) {
      long rr = row0 + v + 8 * half;
      int c = nt * 16 + n;
      float res = (rr < nrows) ? X[rr * 64L + c] : 0.f;
      sF[w][(v + 8 * half) * 64 + c] = oacc[nt][v] + c2_b[c] + res;
    }
  __syncthreads();
  if (lane < 16) {
    float* row = sF[w] + lane * 64;
    float m = 0.f;
#pragma unroll
    for (int j = 0; j < 64; j++) m += row[j];
    m *= (1.f / 64.f);
    float var = 0.f;
#pragma unroll
    for (int j = 0; j < 64; j++) { float d = row[j] - m; var += d * d; }
    var *= (1.f / 64.f);
    sMean[w][lane] = m;
    sInv[w][lane] = rsqrtf(var + 1e-5f);
  }
  __syncthreads();
  for (int i = lane; i < 16 * 64; i += 32) {
    int r = i >> 6, c = i & 63;
    long rr = row0 + r;
    if (rr < nrows)
      out[rr * 64L + c] =
          (sF[w][i] - sMean[w][r]) * sInv[w][r] * ln_g[c] + ln_b[c];
  }
}

// Final projection: [rows,64] @ [64,2] + b
__global__ void k_proj(const float* X, const float* ow, const float* ob,
                       float* out, long nrows) {
  long t = (long)blockIdx.x * blockDim.x + threadIdx.x;
  if (t >= nrows) return;
  const float* x = X + t * 64;
  float s0 = ob[0], s1 = ob[1];
#pragma unroll
  for (int k = 0; k < 64; k++) { s0 += x[k] * ow[k * 2]; s1 += x[k] * ow[k * 2 + 1]; }
  out[t * 2] = s0;
  out[t * 2 + 1] = s1;
}

// ---------------------------------------------------------------------------
// Host orchestration
// ---------------------------------------------------------------------------
// Input flattening assumption: top-level dict insertion order; nested pytrees
// flattened recursively in insertion order:
//  0 x_seq, 1 y_seq, 2 edge_index, 3 edge_rel, 4 xt_idx, 5 yt_idx,
//  6..77   enc_gcn[i][k]{basis,comp,root},
//  78..149 dec_gcn[i][k]{basis,comp,root},
//  150..162 enc_layers[0]: attn{Wq,Wk,Wv,fc_w,fc_b}, ff{c1_w,c1_b,c2_w,c2_b},
//           n1{g,b}, n2{g,b},
//  163..182 dec_layers[0]: self{..5}, cross{..5}, ff{..4}, n1{g,b}, n2{g,b},
//           n3{g,b},
//  183 out_w, 184 out_b.

extern "C" void kernel_launch(void* const* d_in, const int* in_sizes, int n_in,
                              void* d_out, int out_size, void* d_ws,
                              size_t ws_size, hipStream_t stream) {
  (void)in_sizes; (void)n_in; (void)out_size; (void)ws_size;
  const float* x_seq = (const float*)d_in[0];
  const float* y_seq = (const float*)d_in[1];
  const int* e_src = (const int*)d_in[2];
  const int* e_dst = ((const int*)d_in[2]) + NEDGE;
  const int* e_rel = (const int*)d_in[3];
  const int* xt_idx = (const int*)d_in[4];
  const int* yt_idx = (const int*)d_in[5];
  auto F = [&](int i) { return (const float*)d_in[i]; };

  // workspace layout
  char* base = (char*)d_ws;
  size_t off = 0;
  auto alloc = [&](size_t bytes) {
    size_t r = off;
    off = (off + bytes + 255) & ~(size_t)255;
    return r;
  };
  float* pe = (float*)(base + alloc(MAXLEN * DM * 4));
  float* w0 = (float*)(base + alloc((size_t)2 * H_STEPS * NREL * 2 * DM * 4));
  float* w1 = (float*)(base + alloc((size_t)2 * H_STEPS * NREL * DM * DM * 4));
  __bf16* pkw1 =
      (__bf16*)(base + alloc((size_t)2 * H_STEPS * (NREL + 1) * 4096 * 2));
  __bf16* pkattn = (__bf16*)(base + alloc((size_t)12 * 4096 * 2));
  __bf16* pkc1 = (__bf16*)(base + alloc((size_t)2 * 16384 * 2));
  __bf16* pkc2 = (__bf16*)(base + alloc((size_t)2 * 16384 * 2));
  float* tmp_y = (float*)(base + alloc((size_t)NBT * DM * 4));
  float* tmp_h = (float*)(base + alloc((size_t)NREL * NBT * DM * 4));
  float* bufE = (float*)(base + alloc((size_t)H_STEPS * NBT * DM * 4));
  float* bufD = (float*)(base + alloc((size_t)H_STEPS * NBT * DM * 4));

  k_pe<<<(MAXLEN * DM + 255) / 256, 256, 0, stream>>>(pe);

  // pack transformer weights (12 x 64x64 + 2 ff pairs)
  const int attn_idx[12] = {150, 151, 152, 153,   // enc: Wq Wk Wv fc_w
                            163, 164, 165, 166,   // dec self
                            168, 169, 170, 171};  // dec cross
  for (int m = 0; m < 12; m++)
    k_pack<<<1, 256, 0, stream>>>(F(attn_idx[m]), 64, 64, 0,
                                  pkattn + (size_t)m * 4096);
  k_pack<<<4, 256, 0, stream>>>(F(155), 64, 256, 1, pkc1);            // enc c1
  k_pack<<<4, 256, 0, stream>>>(F(157), 256, 64, 1, pkc2);            // enc c2
  k_pack<<<4, 256, 0, stream>>>(F(173), 64, 256, 1, pkc1 + 16384);    // dec c1
  k_pack<<<4, 256, 0, stream>>>(F(175), 256, 64, 1, pkc2 + 16384);    // dec c2

  const long rowsAll = (long)H_STEPS * NBT;
  const int gBlk = (NBT + 95) / 96;                   // 417
  const int lBlk = (int)((rowsAll + 95) / 96);        // 5000
  const int eBlk = (int)(((long)NEDGE * DM + 255) / 256);

  // RGCN streams (enc: s=0, dec: s=1)
  for (int s = 0; s < 2; s++) {
    const float* seq = s == 0 ? x_seq : y_seq;
    const int* tix = s == 0 ? xt_idx : yt_idx;
    float* buf = s == 0 ? bufE : bufD;
    int gb = s == 0 ? 6 : 78;
    for (int i = 0; i < H_STEPS; i++) {
      const float* basis0 = F(gb + (i * 2 + 0) * 3 + 0);
      const float* comp0 = F(gb + (i * 2 + 0) * 3 + 1);
      const float* root0 = F(gb + (i * 2 + 0) * 3 + 2);
      const float* basis1 = F(gb + (i * 2 + 1) * 3 + 0);
      const float* comp1 = F(gb + (i * 2 + 1) * 3 + 1);
      const float* root1 = F(gb + (i * 2 + 1) * 3 + 2);
      float* w0p = w0 + (size_t)(s * H_STEPS + i) * NREL * 2 * DM;
      float* w1p = w1 + (size_t)(s * H_STEPS + i) * NREL * DM * DM;
      __bf16* pk = pkw1 + (size_t)(s * H_STEPS + i) * (NREL + 1) * 4096;

      k_combine<<<(NREL * 2 * DM + 255) / 256, 256, 0, stream>>>(basis0, comp0,
                                                                 2, w0p);
      k_combine<<<(NREL * DM * DM + 255) / 256, 256, 0, stream>>>(basis1, comp1,
                                                                  DM, w1p);
      for (int r = 0; r < NREL; r++)
        k_pack<<<1, 256, 0, stream>>>(w1p + (size_t)r * DM * DM, 64, 64, 0,
                                      pk + (size_t)r * 4096);
      k_pack<<<1, 256, 0, stream>>>(root1, 64, 64, 0, pk + (size_t)NREL * 4096);

      const float* xi = seq + (size_t)i * NBT * 2;
      float* bi = buf + (size_t)i * NBT * DM;
      k_hop0_root<<<(NBT * DM + 255) / 256, 256, 0, stream>>>(xi, root0, tmp_y);
      k_hop0_edges<<<eBlk, 256, 0, stream>>>(xi, e_src, e_dst, e_rel, w0p,
                                             tmp_y);
      for (int r = 0; r < NREL; r++)
        k_gemm64<<<gBlk, 192, 0, stream>>>(tmp_y, NBT, pk + (size_t)r * 4096,
                                           tmp_h + (size_t)r * NBT * DM, 1,
                                           nullptr, nullptr, 0);
      k_gemm64<<<gBlk, 192, 0, stream>>>(tmp_y, NBT, pk + (size_t)NREL * 4096,
                                         bi, 1, pe, tix, i);
      k_hop1_edges<<<eBlk, 256, 0, stream>>>(e_src, e_dst, e_rel, tmp_h, bi);
    }
  }

  // Encoder layer
  k_attn<<<lBlk, 192, 0, stream>>>(bufE, bufE, rowsAll, pkattn + 0 * 4096,
                                   pkattn + 1 * 4096, pkattn + 2 * 4096,
                                   pkattn + 3 * 4096, F(154), F(159), F(160),
                                   bufE);
  k_ff<<<lBlk, 192, 0, stream>>>(bufE, rowsAll, pkc1, pkc2, F(156), F(158),
                                 F(161), F(162), bufE);

  // Decoder layer
  k_attn<<<lBlk, 192, 0, stream>>>(bufD, bufD, rowsAll, pkattn + 4 * 4096,
                                   pkattn + 5 * 4096, pkattn + 6 * 4096,
                                   pkattn + 7 * 4096, F(167), F(177), F(178),
                                   bufD);
  k_attn<<<lBlk, 192, 0, stream>>>(bufD, bufE, rowsAll, pkattn + 8 * 4096,
                                   pkattn + 9 * 4096, pkattn + 10 * 4096,
                                   pkattn + 11 * 4096, F(172), F(179), F(180),
                                   bufD);
  k_ff<<<lBlk, 192, 0, stream>>>(bufD, rowsAll, pkc1 + 16384, pkc2 + 16384,
                                 F(174), F(176), F(181), F(182), bufD);

  k_proj<<<(int)((rowsAll + 255) / 256), 256, 0, stream>>>(
      bufD, F(183), F(184), (float*)d_out, rowsAll);
}
